// EdgeUpdate_1778116461096
// MI455X (gfx1250) — matrix operand order repaired and checked by reading, not compiled, not run on hardware
//
#include <hip/hip_runtime.h>

typedef __attribute__((ext_vector_type(16))) __bf16 v16bf;
typedef __attribute__((ext_vector_type(8)))  __bf16 v8bf;
typedef __attribute__((ext_vector_type(8)))  float  v8f;
typedef __attribute__((ext_vector_type(4)))  float  v4f;

#define TILE_E 64
#define HN     64
#define HE     128
#define IN_DIM 256
#define W2_OFF (IN_DIM * HE)          // offset of W2 inside bf16 weight buffer
#define WBF_ELEMS (IN_DIM * HE + HE * HE)   // 49152 bf16 elements (96 KB)
#define LN_EPS 1e-5f

// ---- B-fragment helpers (lane%16 = col N, elem q = K row q (q<8) / q+8 (q>=8), stride HE) ----
__device__ __forceinline__ v16bf cvt_row16(const float* __restrict__ p) {
    v16bf r;
#pragma unroll
    for (int q = 0; q < 16; ++q) r[q] = (__bf16)p[(q < 8 ? q : q + 8) * HE];
    return r;
}

__device__ __forceinline__ v16bf load16bf(const __bf16* __restrict__ p) {
    v16bf o;
#pragma unroll
    for (int q = 0; q < 16; ++q) o[q] = p[(q < 8 ? q : q + 8) * HE];
    return o;
}

// ---- A-fragment from LDS: [0..7] = K chunk at base, [8..15] = chunk at base+16 ----
__device__ __forceinline__ v16bf make_afrag(const __bf16* base) {
    const v8bf* p = (const v8bf*)base;
    v8bf c0 = p[0];
    v8bf c1 = p[2];   // +16 bf16 elements
    v16bf a;
#pragma unroll
    for (int q = 0; q < 8; ++q) { a[q] = c0[q]; a[q + 8] = c1[q]; }
    return a;
}

// One-time fp32 -> bf16 weight repack into workspace (idempotent, graph-safe).
__global__ void convert_weights_kernel(const float* __restrict__ W1,
                                       const float* __restrict__ W2,
                                       __bf16* __restrict__ wbf) {
    int i = blockIdx.x * blockDim.x + threadIdx.x;
    if (i < W2_OFF)            wbf[i] = (__bf16)W1[i];
    else if (i < WBF_ELEMS)    wbf[i] = (__bf16)W2[i - W2_OFF];
}

template <bool BW>   // BW: bf16 weights pre-packed in wbf
__global__ __launch_bounds__(256) void edge_mlp_kernel(
    const float* __restrict__ node,      // [N,64]
    const float* __restrict__ edgef,     // [E,128]
    const long long* __restrict__ eidx,  // [2,E]
    const float* __restrict__ W1,        // [256,128] row-major (K,N)
    const float* __restrict__ b1,        // [128]
    const float* __restrict__ W2,        // [128,128]
    const float* __restrict__ b2,        // [128]
    const float* __restrict__ gamma_,    // [128]
    const float* __restrict__ beta_,     // [128]
    const __bf16* __restrict__ wbf,      // bf16 W1|W2 (if BW)
    float* __restrict__ out,             // [E,128]
    long long E)
{
    // sA (64x256 bf16 = 32KB) aliased with sU (64x128 f32 = 32KB); sH = 16KB.
    __shared__ alignas(16) unsigned char smem[TILE_E * IN_DIM * 2 + TILE_E * HE * 2];
    __bf16* sA = (__bf16*)smem;
    float*  sU = (float*)smem;
    __bf16* sH = (__bf16*)(smem + TILE_E * IN_DIM * 2);

    const int tid  = threadIdx.x;
    const int lane = tid & 31;
    const int wave = tid >> 5;              // 0..7 -> N-tile
    const long long e0 = (long long)blockIdx.x * TILE_E;

    // ---------- stage edge features as bf16 into A cols [128..255] ----------
#pragma unroll
    for (int it = 0; it < (TILE_E * HE) / (4 * 256); ++it) {
        int idx = tid + it * 256;
        int lin = idx * 4;
        int e = lin >> 7, c = lin & 127;
        long long ge = e0 + e; if (ge >= E) ge = E - 1;
        v4f v = *(const v4f*)(edgef + ge * HE + c);
        __bf16* pa = &sA[e * IN_DIM + HE + c];
#pragma unroll
        for (int j = 0; j < 4; ++j) pa[j] = (__bf16)v[j];
    }

    // Prefetch next block's edge rows.
    if (e0 + TILE_E < E)
        __builtin_prefetch(edgef + (e0 + TILE_E) * HE + (long long)tid * 32, 0, 1);

    // ---------- gather node features: 4 threads/edge, 16 floats each ----------
    {
        const int e = tid >> 2, j = tid & 3;
        long long ge = e0 + e; if (ge >= E) ge = E - 1;
        const long long si = eidx[ge];
        const long long di = eidx[E + ge];
        const v4f* sp = (const v4f*)(node + si * HN + j * 16);
        const v4f* dp = (const v4f*)(node + di * HN + j * 16);
        __bf16* ps = &sA[e * IN_DIM + j * 16];
        __bf16* pd = &sA[e * IN_DIM + HN + j * 16];
#pragma unroll
        for (int q4 = 0; q4 < 4; ++q4) {
            v4f s = sp[q4], d = dp[q4];
#pragma unroll
            for (int c = 0; c < 4; ++c) {
                ps[q4 * 4 + c] = (__bf16)s[c];
                pd[q4 * 4 + c] = (__bf16)d[c];
            }
        }
    }
    __syncthreads();

    const int nbase = wave * 16;
    const int mrow  = lane & 15;
    const int khalf = lane >> 4;

    // ================= GEMM1: [64x256] x [256x128] =================
    v8f acc[4];
#pragma unroll
    for (int m = 0; m < 4; ++m)
#pragma unroll
        for (int i = 0; i < 8; ++i) acc[m][i] = 0.f;

#pragma unroll
    for (int kk = 0; kk < 8; ++kk) {
        v16bf bfrag = BW ? load16bf(wbf + (kk * 32 + khalf * 8) * HE + nbase + mrow)
                         : cvt_row16(W1 + (kk * 32 + khalf * 8) * HE + nbase + mrow);
#pragma unroll
        for (int m = 0; m < 4; ++m) {
            v16bf a = make_afrag(&sA[(m * 16 + mrow) * IN_DIM + kk * 32 + khalf * 8]);
            acc[m] = __builtin_amdgcn_wmma_f32_16x16x32_bf16(
                false, a, false, bfrag, (short)0, acc[m], false, false);
        }
    }

    // bias + ReLU -> bf16 h tile (C/D layout: VGPR i -> row i + 8*khalf)
    {
        const float bias1 = b1[nbase + mrow];
#pragma unroll
        for (int m = 0; m < 4; ++m)
#pragma unroll
            for (int i = 0; i < 8; ++i) {
                float h = acc[m][i] + bias1;
                h = h > 0.f ? h : 0.f;
                sH[(m * 16 + i + khalf * 8) * HE + nbase + mrow] = (__bf16)h;
            }
    }
    __syncthreads();   // all sA reads done; sU may now overwrite the aliased region

    // ================= GEMM2: [64x128] x [128x128] =================
    v8f upd[4];
#pragma unroll
    for (int m = 0; m < 4; ++m)
#pragma unroll
        for (int i = 0; i < 8; ++i) upd[m][i] = 0.f;

#pragma unroll
    for (int kk = 0; kk < 4; ++kk) {
        v16bf bfrag = BW ? load16bf(wbf + W2_OFF + (kk * 32 + khalf * 8) * HE + nbase + mrow)
                         : cvt_row16(W2 + (kk * 32 + khalf * 8) * HE + nbase + mrow);
#pragma unroll
        for (int m = 0; m < 4; ++m) {
            v16bf a = make_afrag(&sH[(m * 16 + mrow) * HE + kk * 32 + khalf * 8]);
            upd[m] = __builtin_amdgcn_wmma_f32_16x16x32_bf16(
                false, a, false, bfrag, (short)0, upd[m], false, false);
        }
    }

    // bias2 -> fp32 update tile in LDS (aliases dead sA)
    {
        const float bias2 = b2[nbase + mrow];
#pragma unroll
        for (int m = 0; m < 4; ++m)
#pragma unroll
            for (int i = 0; i < 8; ++i)
                sU[(m * 16 + i + khalf * 8) * HE + nbase + mrow] = upd[m][i] + bias2;
    }
    __syncthreads();

    // ================= LayerNorm + residual + store =================
    {
        const int e = tid >> 2;      // 0..63
        const int j = tid & 3;       // 32-feature slice
        const float* u = &sU[e * HE + j * 32];
        float vbuf[32];
        float s = 0.f, s2 = 0.f;
#pragma unroll
        for (int q = 0; q < 32; ++q) {
            float x = u[q];
            vbuf[q] = x; s += x; s2 += x * x;
        }
#pragma unroll
        for (int m = 1; m < 4; m <<= 1) {
            s  += __shfl_xor(s,  m, 32);
            s2 += __shfl_xor(s2, m, 32);
        }
        const float mu   = s  * (1.f / 128.f);
        const float var  = s2 * (1.f / 128.f) - mu * mu;
        const float rinv = rsqrtf(var + LN_EPS);

        const long long ge = e0 + e;
        if (ge < E) {
            const v4f* rr = (const v4f*)(edgef + ge * HE + j * 32);  // residual (L2-hot)
            const float* g  = gamma_ + j * 32;
            const float* bb = beta_  + j * 32;
            float* o = out + ge * HE + j * 32;
#pragma unroll
            for (int q4 = 0; q4 < 8; ++q4) {
                v4f r = rr[q4];
                v4f ov;
#pragma unroll
                for (int c = 0; c < 4; ++c) {
                    int q = q4 * 4 + c;
                    ov[c] = r[c] + (vbuf[q] - mu) * rinv * g[q] + bb[q];
                }
                *(v4f*)(o + q4 * 4) = ov;
            }
        }
    }
}

extern "C" void kernel_launch(void* const* d_in, const int* in_sizes, int n_in,
                              void* d_out, int out_size, void* d_ws, size_t ws_size,
                              hipStream_t stream) {
    const float*     node  = (const float*)d_in[0];
    const float*     edgef = (const float*)d_in[1];
    const long long* eidx  = (const long long*)d_in[2];
    const float*     W1    = (const float*)d_in[3];
    const float*     b1    = (const float*)d_in[4];
    const float*     W2    = (const float*)d_in[5];
    const float*     b2    = (const float*)d_in[6];
    const float*     g     = (const float*)d_in[7];
    const float*     bt    = (const float*)d_in[8];
    float*           out   = (float*)d_out;

    const long long E = (long long)in_sizes[2] / 2;   // edge_index is [2,E]
    const int blocks = (int)((E + TILE_E - 1) / TILE_E);

    if (ws_size >= (size_t)WBF_ELEMS * sizeof(__bf16)) {
        __bf16* wbf = (__bf16*)d_ws;
        convert_weights_kernel<<<(WBF_ELEMS + 255) / 256, 256, 0, stream>>>(W1, W2, wbf);
        edge_mlp_kernel<true><<<blocks, 256, 0, stream>>>(
            node, edgef, eidx, W1, b1, W2, b2, g, bt, wbf, out, E);
    } else {
        edge_mlp_kernel<false><<<blocks, 256, 0, stream>>>(
            node, edgef, eidx, W1, b1, W2, b2, g, bt, nullptr, out, E);
    }
}
